// CNN_LSTM_CTC_29652454211587
// MI455X (gfx1250) — compile-verified, matrix operand order checked by reading, not run on hardware
//
#include <hip/hip_runtime.h>

// ---------------------------------------------------------------------------
// CNN + 5x alternating-direction LSTM (H=384, T=1600, B=128) + dense(384->5)
// CDNA5 (gfx1250) wave32 WMMA bf16 implementation.
// ---------------------------------------------------------------------------

typedef __attribute__((ext_vector_type(16))) __bf16 v16bf;
typedef __attribute__((ext_vector_type(8)))  float  v8f;

#define H      384
#define NG     1536          // 4*H
#define KC     768           // K of fused [x|h] GEMM
#define TT     1600          // time steps after stride-5 conv
#define T1     8000
#define BATCH  128
#define KT_L   24            // 768/32 k-tiles (lstm)
#define NT_L   96            // 1536/16 n-tiles (lstm)
#define KT_C   10            // conv3: K=304 padded to 320 -> 10 k-tiles
#define NT_C   24            // conv3: 384/16 n-tiles

__device__ __forceinline__ v8f wmma_bf16(v16bf a, v16bf b, v8f c) {
  return __builtin_amdgcn_wmma_f32_16x16x32_bf16(
      /*neg_a=*/false, a, /*neg_b=*/false, b,
      /*c_mod=*/(short)0, c, /*reuse_a=*/false, /*reuse_b=*/false);
}

__device__ __forceinline__ float sigmoidf_(float x) {
  return 1.0f / (1.0f + __expf(-x));
}

// K interleave pattern of the ISA 16-bit fragment layout:
// slot s (0..15): v=s/2, odd=s&1, k = ((v<4)?2v:2v+8) + 8*hi + odd
__device__ __forceinline__ int kpat(int slot, int hi) {
  const int v = slot >> 1, odd = slot & 1;
  return ((v < 4) ? 2 * v : 2 * v + 8) + 8 * hi + odd;
}

// ---------------------------------------------------------------------------
// Weight pre-pack kernels (f32 -> bf16, WMMA B-fragment tile layout:
// [ntile][ktile][lane(32)][slot(16)], lane%16 = N column, lane/16 = hi)
// ---------------------------------------------------------------------------
__global__ void prep_lstm_w(const float* __restrict__ Wi,
                            const float* __restrict__ Wh,
                            __bf16* __restrict__ Wp) {
  const size_t g = (size_t)blockIdx.x * 256 + threadIdx.x;
  if (g >= (size_t)5 * NT_L * KT_L * 512) return;
  const int slot = (int)(g & 15);
  const int lane = (int)((g >> 4) & 31);
  size_t tiles = g >> 9;
  const int kt = (int)(tiles % KT_L); tiles /= KT_L;
  const int nt = (int)(tiles % NT_L);
  const int layer = (int)(tiles / NT_L);
  const int n = lane & 15, hi = lane >> 4;
  const int k = kt * 32 + kpat(slot, hi);         // 0..767
  const int ng = nt * 16 + n;                     // 0..1535
  float val;
  if (k < H) val = Wi[((size_t)layer * H + k) * NG + ng];
  else       val = Wh[((size_t)layer * H + (k - H)) * NG + ng];
  Wp[g] = (__bf16)val;
}

__global__ void prep_conv3_w(const float* __restrict__ W,  // (19,16,384)
                             __bf16* __restrict__ Wp) {
  const int g = blockIdx.x * 256 + threadIdx.x;
  if (g >= NT_C * KT_C * 512) return;
  const int slot = g & 15;
  const int lane = (g >> 4) & 31;
  const int tiles = g >> 9;
  const int kt = tiles % KT_C;
  const int nt = tiles / KT_C;
  const int n = lane & 15, hi = lane >> 4;
  const int k = kt * 32 + kpat(slot, hi);         // 0..319 (304 valid)
  float val = 0.0f;
  if (k < 19 * 16) val = W[(size_t)k * H + nt * 16 + n];
  Wp[g] = (__bf16)val;
}

// ---------------------------------------------------------------------------
// Fused conv1 (1->4, k5, SAME) + conv2 (4->16, k5, SAME), both SiLU.
// One thread per (b, t). Output bf16 (B, 8000, 16).
// ---------------------------------------------------------------------------
__global__ __launch_bounds__(256)
void conv12_kernel(const float* __restrict__ x,
                   const float* __restrict__ w1, const float* __restrict__ b1,
                   const float* __restrict__ w2, const float* __restrict__ b2,
                   __bf16* __restrict__ out) {
  __shared__ float sw1[20], sb1[4], sw2[320], sb2[16];
  const int tid = threadIdx.x;
  if (tid < 20) sw1[tid] = w1[tid];
  if (tid < 4)  sb1[tid] = b1[tid];
  if (tid < 16) sb2[tid] = b2[tid];
  for (int i = tid; i < 320; i += 256) sw2[i] = w2[i];
  __syncthreads();

  const size_t g = (size_t)blockIdx.x * 256 + tid;   // 128*8000 exact
  const int b = (int)(g / T1), t = (int)(g % T1);

  float xw[9];
#pragma unroll
  for (int j = 0; j < 9; ++j) {
    const int ti = t - 4 + j;
    xw[j] = (ti >= 0 && ti < T1) ? x[(size_t)b * T1 + ti] : 0.0f;
  }
  float y1[5][4];
#pragma unroll
  for (int p = 0; p < 5; ++p) {
#pragma unroll
    for (int c = 0; c < 4; ++c) {
      float s = sb1[c];
#pragma unroll
      for (int u = 0; u < 5; ++u) s += xw[p + u] * sw1[u * 4 + c];
      y1[p][c] = s * sigmoidf_(s);
    }
  }
#pragma unroll
  for (int c2 = 0; c2 < 16; ++c2) {
    float s = sb2[c2];
#pragma unroll
    for (int p = 0; p < 5; ++p)
#pragma unroll
      for (int c1 = 0; c1 < 4; ++c1)
        s += y1[p][c1] * sw2[(p * 4 + c1) * 16 + c2];
    s = s * sigmoidf_(s);
    out[g * 16 + c2] = (__bf16)s;
  }
}

// ---------------------------------------------------------------------------
// conv3 (16->384, k19, stride5, SAME, SiLU) as im2col WMMA GEMM.
// One workgroup = 1 batch x 16 output timesteps; 8 waves x 3 n-tiles.
// ---------------------------------------------------------------------------
__global__ __launch_bounds__(256, 2)
void conv3_kernel(const __bf16* __restrict__ x2, const __bf16* __restrict__ Wp,
                  const float* __restrict__ bias, __bf16* __restrict__ actout) {
  __shared__ __bf16 A[16][320 + 8];
  const int tid = threadIdx.x;
  const int wave = tid >> 5;        // 0..7
  const int lane = tid & 31;
  const int n = lane & 15, hi = lane >> 4;
  const int b = blockIdx.x / 100;
  const int t0 = (blockIdx.x % 100) * 16;

  // im2col stage: row r = local t, col kk = tap*16 + cin (304 valid, pad 320)
  for (int idx = tid; idx < 16 * 320; idx += 256) {
    const int r = idx / 320, kk = idx % 320;
    __bf16 val = (__bf16)0.0f;
    if (kk < 304) {
      const int w = kk >> 4, cin = kk & 15;
      const int t_in = (t0 + r) * 5 + w - 7;      // SAME pad_left = 7
      if (t_in >= 0 && t_in < T1)
        val = x2[((size_t)b * T1 + t_in) * 16 + cin];
    }
    A[r][kk] = val;
  }
  __syncthreads();

  v8f acc0 = {}, acc1 = {}, acc2 = {};
  const int nt0 = wave * 3;
  const size_t nts = (size_t)KT_C * 512;
  for (int kt = 0; kt < KT_C; ++kt) {
    v16bf a;
    const __bf16* ar = &A[lane & 15][kt * 32];
#pragma unroll
    for (int v = 0; v < 8; ++v) {
      const int kb = ((v < 4) ? 2 * v : 2 * v + 8) + 8 * hi;
      a[2 * v]     = ar[kb];
      a[2 * v + 1] = ar[kb + 1];
    }
    const size_t base = ((size_t)kt * 32 + lane) * 16;
    v16bf bA = *(const v16bf*)(Wp + (size_t)(nt0 + 0) * nts + base);
    v16bf bB = *(const v16bf*)(Wp + (size_t)(nt0 + 1) * nts + base);
    v16bf bC = *(const v16bf*)(Wp + (size_t)(nt0 + 2) * nts + base);
    acc0 = wmma_bf16(a, bA, acc0);
    acc1 = wmma_bf16(a, bB, acc1);
    acc2 = wmma_bf16(a, bC, acc2);
  }
#pragma unroll
  for (int j = 0; j < 3; ++j) {
    v8f acc = (j == 0) ? acc0 : ((j == 1) ? acc1 : acc2);
    const int col = (nt0 + j) * 16 + n;
    const float bb = bias[col];
#pragma unroll
    for (int r = 0; r < 8; ++r) {
      const int m = r + 8 * hi;
      float v = acc[r] + bb;
      v = v * sigmoidf_(v);
      actout[((size_t)b * TT + (t0 + m)) * H + col] = (__bf16)v;
    }
  }
}

// ---------------------------------------------------------------------------
// One LSTM layer. Grid = 8 workgroups (batch split, M=16 rows each).
// 24 waves; wave w owns h-columns [16w,16w+16) and gate tiles {w, w+24, w+48,
// w+72} of z = [x_t|h] @ [Wi;Wh] + b. Per step: 24 k-iters x 4 WMMA.
// W streams from L2 (2.25 MB/layer). Two barriers per step.
// ---------------------------------------------------------------------------
__global__ __launch_bounds__(768, 1)
void lstm_kernel(const __bf16* __restrict__ actin, __bf16* __restrict__ actout,
                 const __bf16* __restrict__ WpAll, const float* __restrict__ bias,
                 int layer, int reverse) {
  __shared__ __bf16 A[16][KC + 8];     // [x_t (0:384) | h (384:768)]
  const int tid = threadIdx.x;
  const int wave = tid >> 5;           // 0..23
  const int lane = tid & 31;
  const int n = lane & 15, hi = lane >> 4;
  const int bg = blockIdx.x;           // rows m -> batch bg*16+m

  const __bf16* Wl = WpAll + (size_t)layer * NT_L * KT_L * 512;
  const float*  bl = bias + (size_t)layer * NG;
  const int jc = wave * 16 + n;        // this lane's h column
  const float b_i = bl[jc], b_f = bl[H + jc], b_g = bl[2 * H + jc], b_o = bl[3 * H + jc];
  const size_t nts = (size_t)KT_L * 512;
  const size_t ntI = (size_t)wave * nts;
  const size_t ntF = (size_t)(wave + 24) * nts;
  const size_t ntG = (size_t)(wave + 48) * nts;
  const size_t ntO = (size_t)(wave + 72) * nts;

  // init h = 0 in LDS, c = 0 in registers
  for (int idx = tid; idx < 16 * H; idx += 768)
    A[idx / H][H + (idx % H)] = (__bf16)0.0f;
  v8f cst = {};
  __syncthreads();

  const int sm = tid / 48;             // staging row
  const int sc = (tid % 48) * 8;       // staging col (8 bf16 = 16B)

  for (int it = 0; it < TT; ++it) {
    const int t = reverse ? (TT - 1 - it) : it;

    // stage x_t -> A[:, 0:384] (disjoint from h region; reads of previous
    // iteration finished before the post-GEMM barrier)
    {
      const __bf16* src = actin + ((size_t)(bg * 16 + sm) * TT + t) * H + sc;
      *(float4*)&A[sm][sc] = *(const float4*)src;
      // prefetch next timestep's row
      const int tn = reverse ? (t - 1) : (t + 1);
      if (tn >= 0 && tn < TT)
        __builtin_prefetch(actin + ((size_t)(bg * 16 + sm) * TT + tn) * H + sc, 0, 0);
    }
    __syncthreads();   // x staged + previous h writes visible

    v8f zi = {}, zf = {}, zg = {}, zo = {};
    for (int kt = 0; kt < KT_L; ++kt) {
      v16bf a;
      const __bf16* ar = &A[lane & 15][kt * 32];
#pragma unroll
      for (int v = 0; v < 8; ++v) {
        const int kb = ((v < 4) ? 2 * v : 2 * v + 8) + 8 * hi;
        a[2 * v]     = ar[kb];
        a[2 * v + 1] = ar[kb + 1];
      }
      const size_t base = ((size_t)kt * 32 + lane) * 16;
      v16bf wI = *(const v16bf*)(Wl + ntI + base);
      v16bf wF = *(const v16bf*)(Wl + ntF + base);
      v16bf wG = *(const v16bf*)(Wl + ntG + base);
      v16bf wO = *(const v16bf*)(Wl + ntO + base);
      zi = wmma_bf16(a, wI, zi);
      zf = wmma_bf16(a, wF, zf);
      zg = wmma_bf16(a, wG, zg);
      zo = wmma_bf16(a, wO, zo);
    }
    __syncthreads();   // all LDS A reads done before h region is overwritten

    // gates + state update; write h to LDS (next step) and global (next layer)
#pragma unroll
    for (int r = 0; r < 8; ++r) {
      const float vi = sigmoidf_(zi[r] + b_i);
      const float vf = sigmoidf_(zf[r] + b_f);
      const float vg = tanhf(zg[r] + b_g);
      const float vo = sigmoidf_(zo[r] + b_o);
      const float cn = vf * cst[r] + vi * vg;
      cst[r] = cn;
      const float hv = vo * tanhf(cn);
      const int m = r + 8 * hi;
      const __bf16 hb = (__bf16)hv;
      A[m][H + jc] = hb;
      actout[((size_t)(bg * 16 + m) * TT + t) * H + jc] = hb;
    }
  }
}

// ---------------------------------------------------------------------------
// Dense 384 -> 5 (+bias), f32 output. One thread per (b, t).
// ---------------------------------------------------------------------------
__global__ __launch_bounds__(256)
void dense_kernel(const __bf16* __restrict__ act, const float* __restrict__ dw,
                  const float* __restrict__ db, float* __restrict__ out) {
  __shared__ float sw[H * 5];
  __shared__ float sb[5];
  const int tid = threadIdx.x;
  for (int i = tid; i < H * 5; i += 256) sw[i] = dw[i];
  if (tid < 5) sb[tid] = db[tid];
  __syncthreads();

  const size_t g = (size_t)blockIdx.x * 256 + tid;   // 128*1600 exact
  const __bf16* row = act + g * H;
  float acc[5] = {sb[0], sb[1], sb[2], sb[3], sb[4]};
  for (int c = 0; c < H; ++c) {
    const float a = (float)row[c];
#pragma unroll
    for (int o = 0; o < 5; ++o) acc[o] += a * sw[c * 5 + o];
  }
#pragma unroll
  for (int o = 0; o < 5; ++o) out[g * 5 + o] = acc[o];
}

// ---------------------------------------------------------------------------
extern "C" void kernel_launch(void* const* d_in, const int* in_sizes, int n_in,
                              void* d_out, int out_size, void* d_ws, size_t ws_size,
                              hipStream_t stream) {
  const float* x       = (const float*)d_in[0];
  const float* conv1_w = (const float*)d_in[1];
  const float* conv1_b = (const float*)d_in[2];
  const float* conv2_w = (const float*)d_in[3];
  const float* conv2_b = (const float*)d_in[4];
  const float* conv3_w = (const float*)d_in[5];
  const float* conv3_b = (const float*)d_in[6];
  const float* lstm_Wi = (const float*)d_in[7];
  const float* lstm_Wh = (const float*)d_in[8];
  const float* lstm_b  = (const float*)d_in[9];
  const float* dense_w = (const float*)d_in[10];
  const float* dense_b = (const float*)d_in[11];

  char* ws = (char*)d_ws;
  size_t off = 0;
  auto wsalloc = [&](size_t bytes) -> void* {
    void* p = ws + off;
    off = (off + bytes + 255) & ~(size_t)255;
    return p;
  };
  __bf16* conv2out = (__bf16*)wsalloc((size_t)BATCH * T1 * 16 * 2);        // 32.8 MB
  __bf16* act0     = (__bf16*)wsalloc((size_t)BATCH * TT * H * 2);         // 157 MB
  __bf16* act1     = (__bf16*)wsalloc((size_t)BATCH * TT * H * 2);         // 157 MB
  __bf16* WpL      = (__bf16*)wsalloc((size_t)5 * NT_L * KT_L * 512 * 2);  // 11.8 MB
  __bf16* WpC      = (__bf16*)wsalloc((size_t)NT_C * KT_C * 512 * 2);      // 0.25 MB
  (void)ws_size; (void)in_sizes; (void)n_in; (void)out_size;

  prep_conv3_w<<<(NT_C * KT_C * 512 + 255) / 256, 256, 0, stream>>>(conv3_w, WpC);
  prep_lstm_w<<<(int)(((size_t)5 * NT_L * KT_L * 512 + 255) / 256), 256, 0, stream>>>(
      lstm_Wi, lstm_Wh, WpL);

  conv12_kernel<<<(BATCH * T1) / 256, 256, 0, stream>>>(
      x, conv1_w, conv1_b, conv2_w, conv2_b, conv2out);

  conv3_kernel<<<BATCH * 100, 256, 0, stream>>>(conv2out, WpC, conv3_b, act0);

  __bf16* bufs[2] = {act0, act1};
  for (int i = 0; i < 5; ++i) {
    lstm_kernel<<<BATCH / 16, 768, 0, stream>>>(
        bufs[i % 2], bufs[(i + 1) % 2], WpL, lstm_b, i, (i % 2 == 0) ? 1 : 0);
  }
  // after 5 layers the result lives in bufs[1]
  dense_kernel<<<(BATCH * TT) / 256, 256, 0, stream>>>(bufs[1], dense_w, dense_b,
                                                       (float*)d_out);
}
